// HTGTLayer_71116068487908
// MI455X (gfx1250) — matrix operand order repaired and checked by reading, not compiled, not run on hardware
//
#include <hip/hip_runtime.h>
#include <hip/hip_bf16.h>
#include <math.h>

// HGT-style typed graph attention layer for MI455X (gfx1250, wave32).
// Pipeline: [init] -> [weights->f16 transposed] -> [etype histogram] ->
// [padded offsets] -> [scatter sort] -> [edge tiles: LN + WMMA QKV + logits]
// -> [segment softmax accumulate] -> [dst tiles: WMMA typed output + gate].

typedef __attribute__((ext_vector_type(16))) _Float16 v16h;
typedef __attribute__((ext_vector_type(8)))  float    v8f;

#define ID   128   // in_dim
#define ED   32    // e_dim
#define TD   32    // time dim
#define OD   128   // out_dim
#define KS   160   // in_dim + e_dim
#define H_   8     // heads (head size 16 == WMMA N tile)
#define TILE 16

__device__ __forceinline__ float hsum16(float v) {
  v += __shfl_xor(v, 1, 32);
  v += __shfl_xor(v, 2, 32);
  v += __shfl_xor(v, 4, 32);
  v += __shfl_xor(v, 8, 32);
  return v;
}

// A fragment (16x32 f16, MxK): lanes 0-15 row M=lane holds K {kb..kb+7, kb+16..kb+23},
// lanes 16-31 hold K {kb+8..kb+15, kb+24..kb+31}. p points at [m][kb + (hi?8:0)].
__device__ __forceinline__ v16h load_afrag(const _Float16* p) {
  union { v16h h; uint4 u[2]; } f;
  f.u[0] = *reinterpret_cast<const uint4*>(p);
  f.u[1] = *reinterpret_cast<const uint4*>(p + 16);
  return f.h;
}
// B fragment (32x16 f16, KxN): lane holds column n=lane&15, K {kb..kb+7, kb+16..kb+23}
// for lanes 0-15, {kb+8..kb+15, kb+24..kb+31} for lanes 16-31 (weights stored [n][k]).
__device__ __forceinline__ v16h load_bfrag(const _Float16* p) {
  union { v16h h; uint4 u[2]; } f;
  f.u[0] = *reinterpret_cast<const uint4*>(p);
  f.u[1] = *reinterpret_cast<const uint4*>(p + 16);
  return f.h;
}

#define WMMA_F16(a, b, c) \
  __builtin_amdgcn_wmma_f32_16x16x32_f16(false, (a), false, (b), (short)0, (c), false, false)

// ---------------- init ----------------
__global__ void k_init(float* amax, float* denom, float* numer,
                       int* cnt, int* cursor, int* sorted,
                       int M, int Epad, int R) {
  int g = blockIdx.x * 256 + threadIdx.x;
  if (g < M * H_) { amax[g] = -INFINITY; denom[g] = 0.f; }
  if (g < M * OD) numer[g] = 0.f;
  if (g < Epad)   sorted[g] = -1;
  if (g < R)      { cnt[g] = 0; cursor[g] = 0; }
}

// ---------------- weight convert: f32 [r][k][n] -> f16 [r][n][k] ----------------
__global__ void k_wcvt(const float* __restrict__ Wq, const float* __restrict__ Wk,
                       const float* __restrict__ Wv, const float* __restrict__ Wa,
                       _Float16* __restrict__ wqT, _Float16* __restrict__ wkT,
                       _Float16* __restrict__ wvT, _Float16* __restrict__ waT,
                       int R, int T) {
  int g = blockIdx.x * 256 + threadIdx.x;
  const int nq = R * ID * OD;      // wqT elems
  const int nk = R * OD * KS;      // wkT elems (== wvT)
  const int na = T * OD * OD;
  if (g < nq) {
    int r = g / (OD * ID), rem = g % (OD * ID), n = rem / ID, k = rem % ID;
    wqT[g] = (_Float16)Wq[((size_t)r * ID + k) * OD + n];
  } else if (g < nq + 2 * nk) {
    int gg = g - nq; int which = gg / nk; gg %= nk;
    int r = gg / (OD * KS), rem = gg % (OD * KS), n = rem / KS, k = rem % KS;
    float v = (which ? Wv : Wk)[((size_t)r * KS + k) * OD + n];
    (which ? wvT : wkT)[gg] = (_Float16)v;
  } else if (g < nq + 2 * nk + na) {
    int gg = g - nq - 2 * nk;
    int t = gg / (OD * OD), rem = gg % (OD * OD), n = rem / OD, k = rem % OD;
    waT[gg] = (_Float16)Wa[((size_t)t * OD + k) * OD + n];
  }
}

// ---------------- sort edges by etype ----------------
__global__ void k_hist(const int* __restrict__ etype, int* cnt, int E) {
  int g = blockIdx.x * 256 + threadIdx.x;
  if (g < E) atomicAdd(&cnt[etype[g]], 1);
}
__global__ void k_offsets(const int* __restrict__ cnt, int* off, int R) {
  if (blockIdx.x == 0 && threadIdx.x == 0) {
    int o = 0;
    for (int r = 0; r < R; ++r) { off[r] = o; o += ((cnt[r] + TILE - 1) / TILE) * TILE; }
    off[R] = o;
  }
}
__global__ void k_scatter(const int* __restrict__ etype, const int* __restrict__ off,
                          int* cursor, int* sorted, int E) {
  int g = blockIdx.x * 256 + threadIdx.x;
  if (g < E) {
    int r = etype[g];
    int pos = atomicAdd(&cursor[r], 1);
    sorted[off[r] + pos] = g;
  }
}

// ---------------- edge tiles: temporal enc + LN + WMMA QKV + logits ----------------
__global__ void __launch_bounds__(256)
k_edge(const float* __restrict__ src_h, const float* __restrict__ src_tw,
       const float* __restrict__ src_tb, const float* __restrict__ edge_h,
       const float* __restrict__ date, const int* __restrict__ src_idx,
       const int* __restrict__ dst_idx,
       const float* __restrict__ sg, const float* __restrict__ sb,
       const float* __restrict__ dg, const float* __restrict__ db,
       const _Float16* __restrict__ wqT, const _Float16* __restrict__ wkT,
       const _Float16* __restrict__ wvT,
       const int* __restrict__ off, const int* __restrict__ sorted,
       float* __restrict__ abuf, _Float16* __restrict__ vbuf,
       float* __restrict__ amax, int R) {
  __shared__ _Float16 sAd[TILE][ID];
  __shared__ _Float16 sAs[TILE][KS];
  __shared__ int sEid[TILE];
  __shared__ int sDst[TILE];

  const int base = blockIdx.x * TILE;
  if (base >= off[R]) return;
  int r = 0;
  while (r + 1 < R && base >= off[r + 1]) ++r;   // tile's edge type

  const int tid = threadIdx.x;
  const int m  = tid >> 4;      // row (edge in tile), one 16-lane half per row
  const int ln = tid & 15;

  const int eid = sorted[base + m];
  int si = 0, di = 0; float tt = 0.f;
  if (eid >= 0) { si = src_idx[eid]; di = dst_idx[eid]; tt = date[eid]; }
  if (ln == 0) { sEid[m] = eid; sDst[m] = di; }

  // ---- dia_d (128) : temporal encode + LayerNorm -> f16 LDS ----
  float vd[8]; float sum = 0.f, ss = 0.f;
  #pragma unroll
  for (int j = 0; j < 8; ++j) {
    int c = ln + 16 * j;
    float x = 0.f;
    if (eid >= 0) {
      float hv = src_h[(size_t)di * ID + c];
      x = (c < TD) ? __sinf(src_tw[(size_t)di * TD + c] * tt + src_tb[(size_t)di * TD + c]) * hv
                   : hv;
    }
    vd[j] = x; sum += x; ss += x * x;
  }
  sum = hsum16(sum); ss = hsum16(ss);
  {
    float mu = sum * (1.f / ID);
    float rs = rsqrtf(ss * (1.f / ID) - mu * mu + 1e-5f);
    #pragma unroll
    for (int j = 0; j < 8; ++j) {
      int c = ln + 16 * j;
      float y = (eid >= 0) ? ((vd[j] - mu) * rs * dg[c] + db[c]) : 0.f;
      sAd[m][c] = (_Float16)y;
    }
  }
  // ---- dia_s (160 = src||edge) : temporal encode + LayerNorm -> f16 LDS ----
  float vs[10]; sum = 0.f; ss = 0.f;
  #pragma unroll
  for (int j = 0; j < 10; ++j) {
    int c = ln + 16 * j;
    float x = 0.f;
    if (eid >= 0) {
      if (c < ID) {
        float hv = src_h[(size_t)si * ID + c];
        x = (c < TD) ? __sinf(src_tw[(size_t)si * TD + c] * tt + src_tb[(size_t)si * TD + c]) * hv
                     : hv;
      } else {
        x = edge_h[(size_t)eid * ED + (c - ID)];
      }
    }
    vs[j] = x; sum += x; ss += x * x;
  }
  sum = hsum16(sum); ss = hsum16(ss);
  {
    float mu = sum * (1.f / KS);
    float rs = rsqrtf(ss * (1.f / KS) - mu * mu + 1e-5f);
    #pragma unroll
    for (int j = 0; j < 10; ++j) {
      int c = ln + 16 * j;
      float y = (eid >= 0) ? ((vs[j] - mu) * rs * sg[c] + sb[c]) : 0.f;
      sAs[m][c] = (_Float16)y;
    }
  }
  __syncthreads();

  // ---- WMMA: wave w owns output columns [16w,16w+16) == attention head w ----
  const int w    = tid >> 5;
  const int lane = tid & 31;
  const int nl   = lane & 15;
  const int hi   = lane >> 4;
  const int kA   = hi ? 8 : 0;
  const int kB   = hi ? 8 : 0;
  const int col0 = w * 16;

  v8f accq = {}, acck = {}, accv = {};
  const _Float16* wq = wqT + (size_t)r * OD * ID;   // [n][k] 128x128
  const _Float16* wk = wkT + (size_t)r * OD * KS;   // [n][k] 128x160
  const _Float16* wv = wvT + (size_t)r * OD * KS;

  #pragma unroll
  for (int kb = 0; kb < ID; kb += 32) {
    v16h a = load_afrag(&sAd[nl][kb + kA]);
    v16h b = load_bfrag(wq + (size_t)(col0 + nl) * ID + kb + kB);
    accq = WMMA_F16(a, b, accq);
  }
  #pragma unroll
  for (int kb = 0; kb < KS; kb += 32) {
    v16h a  = load_afrag(&sAs[nl][kb + kA]);
    v16h bk = load_bfrag(wk + (size_t)(col0 + nl) * KS + kb + kB);
    acck = WMMA_F16(a, bk, acck);
    v16h bv = load_bfrag(wv + (size_t)(col0 + nl) * KS + kb + kB);
    accv = WMMA_F16(a, bv, accv);
  }

  // ---- per-edge head-w logit = (q.k)/sqrt(128), V tile to scratch ----
  const float sc = 0.088388347648318447f;   // 1/sqrt(128)
  #pragma unroll
  for (int j = 0; j < 8; ++j) {
    float p = accq[j] * acck[j];
    p = hsum16(p);                           // reduce over the 16 head columns
    int mm = j + (hi ? 8 : 0);
    int e  = sEid[mm];
    if (e >= 0) {
      if (nl == 0) {
        float av = p * sc;
        abuf[(size_t)e * H_ + w] = av;
        atomicMax(&amax[(size_t)sDst[mm] * H_ + w], av);
      }
      vbuf[(size_t)e * OD + col0 + nl] = (_Float16)accv[j];
    }
  }
}

// ---------------- segment softmax accumulate ----------------
__global__ void k_soft(const float* __restrict__ abuf, const _Float16* __restrict__ vbuf,
                       const float* __restrict__ amax, const int* __restrict__ dst_idx,
                       float* __restrict__ denom, float* __restrict__ numer, int E) {
  int g = blockIdx.x * 256 + threadIdx.x;
  if (g >= E * H_) return;
  int e = g >> 3, h = g & 7;
  int d = dst_idx[e];
  float ex = __expf(abuf[g] - amax[(size_t)d * H_ + h]);
  atomicAdd(&denom[(size_t)d * H_ + h], ex);
  const _Float16* vp = vbuf + (size_t)e * OD + h * 16;
  float* np = numer + (size_t)d * OD + h * 16;
  #pragma unroll
  for (int c = 0; c < 16; ++c) atomicAdd(&np[c], ex * (float)vp[c]);
}

// ---------------- dst tiles: typed output projection + gated skip ----------------
__global__ void __launch_bounds__(256)
k_out(const float* __restrict__ numer, const float* __restrict__ denom,
      const float* __restrict__ h_bias, const float* __restrict__ skip,
      const int* __restrict__ ntype, const _Float16* __restrict__ waT,
      const float* __restrict__ src_h, float* __restrict__ out, int M, int T) {
  __shared__ _Float16 sH[TILE][OD];
  __shared__ _Float16 sA[TILE][OD];
  __shared__ float sGate[TILE];
  __shared__ int sNt[TILE];

  const int base = blockIdx.x * TILE;
  const int tid = threadIdx.x;
  const int m  = tid >> 4;
  const int ln = tid & 15;
  const int i  = base + m;

  int nt = 0;
  if (i < M) nt = ntype[i];
  if (ln == 0) {
    sNt[m] = nt;
    float sk = (i < M) ? skip[nt] : 0.f;
    sGate[m] = 1.f / (1.f + __expf(-sk));
  }
  #pragma unroll
  for (int j = 0; j < 8; ++j) {
    int c = ln + 16 * j;
    float y = 0.f;
    if (i < M) {
      float den = denom[(size_t)i * H_ + (c >> 4)];
      float nu  = numer[(size_t)i * OD + c];
      y = (den > 0.f ? nu / den : 0.f) + h_bias[(size_t)nt * OD + c];
    }
    sH[m][c] = (_Float16)y;
  }
  __syncthreads();

  const int w = tid >> 5, lane = tid & 31;
  const int nl = lane & 15, hi = lane >> 4;
  const int kA = hi ? 8 : 0, kB = hi ? 8 : 0;
  const int col0 = w * 16;

  // typed_linear over T=4 node types via masked-row WMMA accumulation
  v8f acc = {};
  for (int t = 0; t < T; ++t) {
    #pragma unroll
    for (int j = 0; j < 8; ++j) {
      int c = ln + 16 * j;
      sA[m][c] = (sNt[m] == t) ? sH[m][c] : (_Float16)0.f;
    }
    __syncthreads();
    const _Float16* wa = waT + (size_t)t * OD * OD;
    #pragma unroll
    for (int kb = 0; kb < OD; kb += 32) {
      v16h a = load_afrag(&sA[nl][kb + kA]);
      v16h b = load_bfrag(wa + (size_t)(col0 + nl) * OD + kb + kB);
      acc = WMMA_F16(a, b, acc);
    }
    __syncthreads();
  }

  #pragma unroll
  for (int j = 0; j < 8; ++j) {
    int mm = j + (hi ? 8 : 0);
    int ii = base + mm;
    if (ii < M) {
      int c = col0 + nl;
      float g = sGate[mm];
      out[(size_t)ii * OD + c] = acc[j] * g + src_h[(size_t)ii * OD + c] * (1.f - g);
    }
  }
}

extern "C" void kernel_launch(void* const* d_in, const int* in_sizes, int n_in,
                              void* d_out, int out_size, void* d_ws, size_t ws_size,
                              hipStream_t stream) {
  (void)n_in; (void)out_size; (void)ws_size;
  const float* src_h   = (const float*)d_in[0];
  const float* src_tw  = (const float*)d_in[1];
  const float* src_tb  = (const float*)d_in[2];
  const float* edge_h  = (const float*)d_in[3];
  const float* date    = (const float*)d_in[4];
  const int*   src_idx = (const int*)d_in[5];
  const int*   dst_idx = (const int*)d_in[6];
  const int*   etype   = (const int*)d_in[7];
  const int*   ntype   = (const int*)d_in[8];
  const float* Wq      = (const float*)d_in[9];
  const float* Wk      = (const float*)d_in[10];
  const float* Wv      = (const float*)d_in[11];
  const float* Wa      = (const float*)d_in[12];
  const float* h_bias  = (const float*)d_in[13];
  const float* skip    = (const float*)d_in[14];
  const float* sg      = (const float*)d_in[15];
  const float* sb      = (const float*)d_in[16];
  const float* dg      = (const float*)d_in[17];
  const float* db      = (const float*)d_in[18];

  const int E = in_sizes[4];
  const int M = in_sizes[8];
  const int R = in_sizes[9] / (ID * OD);
  const int T = in_sizes[14];
  const int Epad = E + TILE * R;

  char* ws = (char*)d_ws; size_t o = 0;
  auto take = [&](size_t bytes) -> char* {
    o = (o + 255) & ~(size_t)255;
    char* p = ws + o; o += bytes; return p;
  };
  int*      cnt    = (int*)take((size_t)R * 4);
  int*      cursor = (int*)take((size_t)R * 4);
  int*      off    = (int*)take((size_t)(R + 1) * 4);
  int*      sorted = (int*)take((size_t)Epad * 4);
  float*    amax   = (float*)take((size_t)M * H_ * 4);
  float*    denom  = (float*)take((size_t)M * H_ * 4);
  float*    numer  = (float*)take((size_t)M * OD * 4);
  float*    abuf   = (float*)take((size_t)E * H_ * 4);
  _Float16* vbuf   = (_Float16*)take((size_t)E * OD * 2);
  _Float16* wqT    = (_Float16*)take((size_t)R * OD * ID * 2);
  _Float16* wkT    = (_Float16*)take((size_t)R * OD * KS * 2);
  _Float16* wvT    = (_Float16*)take((size_t)R * OD * KS * 2);
  _Float16* waT    = (_Float16*)take((size_t)T * OD * OD * 2);

  int initN = M * OD;
  if (Epad > initN) initN = Epad;
  if (M * H_ > initN) initN = M * H_;

  dim3 blk(256);
  k_init<<<(initN + 255) / 256, blk, 0, stream>>>(amax, denom, numer, cnt, cursor, sorted,
                                                  M, Epad, R);
  int wTot = R * OD * ID + 2 * R * OD * KS + T * OD * OD;
  k_wcvt<<<(wTot + 255) / 256, blk, 0, stream>>>(Wq, Wk, Wv, Wa, wqT, wkT, wvT, waT, R, T);
  k_hist<<<(E + 255) / 256, blk, 0, stream>>>(etype, cnt, E);
  k_offsets<<<1, 32, 0, stream>>>(cnt, off, R);
  k_scatter<<<(E + 255) / 256, blk, 0, stream>>>(etype, off, cursor, sorted, E);
  k_edge<<<Epad / TILE, blk, 0, stream>>>(src_h, src_tw, src_tb, edge_h, date, src_idx,
                                          dst_idx, sg, sb, dg, db, wqT, wkT, wvT,
                                          off, sorted, abuf, vbuf, amax, R);
  k_soft<<<(E * H_ + 255) / 256, blk, 0, stream>>>(abuf, vbuf, amax, dst_idx, denom, numer, E);
  k_out<<<(M + TILE - 1) / TILE, blk, 0, stream>>>(numer, denom, h_bias, skip, ntype, waT,
                                                   src_h, (float*)d_out, M, T);
}
